// SOA_15719580304430
// MI455X (gfx1250) — matrix operand order repaired and checked by reading, not compiled, not run on hardware
//
#include <hip/hip_runtime.h>

typedef unsigned short U16;
typedef __attribute__((ext_vector_type(8)))  float  v8f;
typedef __attribute__((ext_vector_type(16))) __bf16 v16bf;

#define N_SP   2304   // H*W
#define C_CH   1024
#define MID_CH 512
#define BATCH  16
#define SM_SCALE 0.04419417382415922f  // 512^-0.5

union FragU { uint4 u[2]; v16bf v; };

__device__ __forceinline__ U16 f2bf(float f) {
  unsigned u = __float_as_uint(f);
  u += 0x7fffu + ((u >> 16) & 1u);   // round-to-nearest-even
  return (U16)(u >> 16);
}

// ---------------------------------------------------------------------------
// WMMA GEMM core, double-buffered LDS, BK=32.
//   Workgroup = 256 threads = 8 waves arranged 4(M) x 2(N).
//   Workgroup tile: 128 x (32*NF).  Wave tile: 32 x (16*NF) = 2 x NF frags.
//   A row-major bf16 (lda halfs); B COLUMN-major: Bt[n][k] (ldb halfs).
// One barrier per K-step; next tile's global loads issued before the WMMAs.
// ---------------------------------------------------------------------------
#define TP 40   // LDS row pitch in halfs (32 data + 8 pad)

template <int NF>
__device__ __forceinline__ void gemm_tile(
    const U16* __restrict__ A, int lda,
    const U16* __restrict__ Bt, int ldb,
    int m0, int n0, int K, v8f (&acc)[2][NF])
{
  constexpr int BROWS = 32 * NF;            // B tile rows = workgroup N tile
  __shared__ U16 As[2][128 * TP];
  __shared__ U16 Bs[2][BROWS * TP];

  const int t    = threadIdx.x;
  const int lane = t & 31;
  const int w    = t >> 5;
  const int wm   = w >> 1;      // 0..3  (M offset wm*32)
  const int wn   = w & 1;       // 0..1  (N offset wn*16*NF)
  const int hh   = lane >> 4;
  const int l16  = lane & 15;

  v8f zf = {};
#pragma unroll
  for (int i = 0; i < 2; ++i)
#pragma unroll
    for (int j = 0; j < NF; ++j) acc[i][j] = zf;

  const int ar = t >> 1;            // A tile row 0..127
  const int ac = (t & 1) << 4;      // A tile col 0/16 (16 halfs)
  const size_t a_goff = (size_t)(m0 + ar) * lda + ac;

  int br, bc;
  if constexpr (NF == 4) { br = t >> 1; bc = (t & 1) << 4; }   // 128 rows x 16 halfs
  else                   { br = t >> 2; bc = (t & 3) << 3; }   // 64 rows x 8 halfs
  const size_t b_goff = (size_t)(n0 + br) * ldb + bc;

  uint4* adst0 = (uint4*)&As[0][ar * TP + ac];
  uint4* adst1 = (uint4*)&As[1][ar * TP + ac];
  uint4* bdst0 = (uint4*)&Bs[0][br * TP + bc];
  uint4* bdst1 = (uint4*)&Bs[1][br * TP + bc];

  // ---- prologue: stage tile 0 into buffer 0
  {
    const uint4* sa = (const uint4*)(A + a_goff);
    adst0[0] = sa[0]; adst0[1] = sa[1];
    const uint4* sb = (const uint4*)(Bt + b_goff);
    bdst0[0] = sb[0];
    if constexpr (NF == 4) bdst0[1] = sb[1];
  }
  __syncthreads();

  int cur = 0;
  for (int k0 = 0; k0 < K; k0 += 32) {
    // ---- issue next tile's global loads early (latency hidden by WMMAs)
    uint4 ra0, ra1, rb0, rb1;
    const bool more = (k0 + 32) < K;
    if (more) {
      const uint4* sa = (const uint4*)(A + a_goff + k0 + 32);
      ra0 = sa[0]; ra1 = sa[1];
      const uint4* sb = (const uint4*)(Bt + b_goff + k0 + 32);
      rb0 = sb[0];
      if constexpr (NF == 4) rb1 = sb[1];
    }

    // ---- fragments from current buffer
    // A: lane holds row m=l16; halves i -> K = (i/8)*16 + hh*8 + i%8
    // B: lane holds col n=l16; halves i -> K = (i/8)*16 + hh*8 + i%8
    FragU a[2], b[NF];
    const U16* arow0 = &As[cur][(wm * 32 + l16) * TP];
    const U16* arow1 = &As[cur][(wm * 32 + 16 + l16) * TP];
    a[0].u[0] = *(const uint4*)(arow0 + hh * 8);
    a[0].u[1] = *(const uint4*)(arow0 + 16 + hh * 8);
    a[1].u[0] = *(const uint4*)(arow1 + hh * 8);
    a[1].u[1] = *(const uint4*)(arow1 + 16 + hh * 8);
#pragma unroll
    for (int fc = 0; fc < NF; ++fc) {
      const U16* brow = &Bs[cur][(wn * (16 * NF) + fc * 16 + l16) * TP];
      b[fc].u[0] = *(const uint4*)(brow + hh * 8);
      b[fc].u[1] = *(const uint4*)(brow + 16 + hh * 8);
    }

#pragma unroll
    for (int fc = 0; fc < NF; ++fc) {
      acc[0][fc] = __builtin_amdgcn_wmma_f32_16x16x32_bf16(false, a[0].v, false, b[fc].v, (short)0, acc[0][fc], false, false);
      acc[1][fc] = __builtin_amdgcn_wmma_f32_16x16x32_bf16(false, a[1].v, false, b[fc].v, (short)0, acc[1][fc], false, false);
    }

    // ---- stage next tile into the other buffer; single barrier per step
    if (more) {
      if (cur == 0) {
        adst1[0] = ra0; adst1[1] = ra1;
        bdst1[0] = rb0;
        if constexpr (NF == 4) bdst1[1] = rb1;
      } else {
        adst0[0] = ra0; adst0[1] = ra1;
        bdst0[0] = rb0;
        if constexpr (NF == 4) bdst0[1] = rb1;
      }
      __syncthreads();
      cur ^= 1;
    }
  }
}

// D-fragment coordinates (16x16 f32): row = r + 8*hh (r = VGPR 0..7), col = l16.

// ---------------------------------------------------------------------------
// Pack kernels: f32 -> bf16 conversions (+ transpose of x).
// ---------------------------------------------------------------------------
__global__ void pack_w(const float* __restrict__ Wf, const float* __restrict__ Wg,
                       const float* __restrict__ Wh, const float* __restrict__ Wv,
                       U16* __restrict__ Wq, U16* __restrict__ Wvh)
{
  int idx = blockIdx.x * 256 + threadIdx.x;
  const int nq = 1536 * 1024;
  if (idx < nq) {
    int o = idx >> 10, cc = idx & 1023;
    float v = (o < 512)  ? Wf[o * 1024 + cc]
            : (o < 1024) ? Wg[(o - 512) * 1024 + cc]
                         : Wh[(o - 1024) * 1024 + cc];
    Wq[idx] = f2bf(v);
  } else {
    int j = idx - nq;
    if (j < 1024 * 512) Wvh[j] = f2bf(Wv[j]);
  }
}

// x[b][c][n] f32 -> xT[b][n][c] bf16 (32x32 LDS tile transpose)
__global__ void pack_x(const float* __restrict__ x, U16* __restrict__ xT)
{
  __shared__ float tile[32][33];
  int b = blockIdx.z, c0 = blockIdx.y * 32, n0 = blockIdx.x * 32;
  int tx = threadIdx.x & 31, ty = threadIdx.x >> 5;   // ty 0..7
#pragma unroll
  for (int p = 0; p < 4; ++p) {
    int c = c0 + ty + p * 8;
    tile[ty + p * 8][tx] = x[((size_t)b * C_CH + c) * N_SP + n0 + tx];
  }
  __syncthreads();
#pragma unroll
  for (int p = 0; p < 4; ++p) {
    int n = n0 + ty + p * 8;
    xT[((size_t)b * N_SP + n) * C_CH + c0 + tx] = f2bf(tile[tx][ty + p * 8]);
  }
}

// ---------------------------------------------------------------------------
// Fused QKV GEMM: [1536 x 1024] x xT-tiles -> f (BN+ReLU, stored [b][n][512]),
//                 g (BN+ReLU, [b][n][512]), h (+bias, [b][512][n]).
// ---------------------------------------------------------------------------
__global__ void gemm_qkv(const U16* __restrict__ Wq, const U16* __restrict__ xT,
                         const float* __restrict__ bf_, const float* __restrict__ gf,
                         const float* __restrict__ betaf, const float* __restrict__ mf,
                         const float* __restrict__ vf,
                         const float* __restrict__ bg_, const float* __restrict__ gg,
                         const float* __restrict__ betag, const float* __restrict__ mg,
                         const float* __restrict__ vg,
                         const float* __restrict__ bh_,
                         U16* __restrict__ fT, U16* __restrict__ gT, U16* __restrict__ hb)
{
  int b = blockIdx.z;
  int o0 = blockIdx.y * 128;
  int n0 = blockIdx.x * 128;
  v8f acc[2][4];
  gemm_tile<4>(Wq, C_CH, xT + (size_t)b * N_SP * C_CH, C_CH, o0, n0, C_CH, acc);

  int t = threadIdx.x, lane = t & 31, w = t >> 5;
  int wm = w >> 1, wn = w & 1, hh = lane >> 4, l16 = lane & 15;
  int region = o0 >> 9;   // 0=f, 1=g, 2=h (tile never straddles: 512 % 128 == 0)

#pragma unroll
  for (int fr = 0; fr < 2; ++fr) {
    int obase = o0 + wm * 32 + fr * 16 + hh * 8;
#pragma unroll
    for (int fc = 0; fc < 4; ++fc) {
      int n = n0 + wn * 64 + fc * 16 + l16;
      if (region < 2) {
        const float* bias = region == 0 ? bf_   : bg_;
        const float* gam  = region == 0 ? gf    : gg;
        const float* bet  = region == 0 ? betaf : betag;
        const float* mu   = region == 0 ? mf    : mg;
        const float* va   = region == 0 ? vf    : vg;
        int ob = obase - region * 512;
        U16* dst = (region == 0 ? fT : gT) + ((size_t)b * N_SP + n) * MID_CH + ob;
        unsigned pk[4];
#pragma unroll
        for (int r = 0; r < 8; ++r) {
          int o = ob + r;
          float sc  = gam[o] * rsqrtf(va[o] + 1e-5f);
          float val = (acc[fr][fc][r] + bias[o]) * sc + (bet[o] - mu[o] * sc);
          val = fmaxf(val, 0.0f);
          unsigned bv16 = f2bf(val);
          if (r & 1) pk[r >> 1] |= bv16 << 16; else pk[r >> 1] = bv16;
        }
        *(uint4*)dst = make_uint4(pk[0], pk[1], pk[2], pk[3]);  // 8 consecutive o
      } else {
        int oh = obase - 1024;
#pragma unroll
        for (int r = 0; r < 8; ++r) {
          float val = acc[fr][fc][r] + bh_[oh + r];
          hb[((size_t)b * MID_CH + oh + r) * N_SP + n] = f2bf(val);
        }
      }
    }
  }
}

// ---------------------------------------------------------------------------
// z[i][j] = sum_m fT[i][m] * gT[j][m]   (one batch; z is f32 [2304][2304])
// ---------------------------------------------------------------------------
__global__ void gemm_z(const U16* __restrict__ fT, const U16* __restrict__ gT,
                       float* __restrict__ z, int b)
{
  int i0 = blockIdx.y * 128, j0 = blockIdx.x * 128;
  v8f acc[2][4];
  gemm_tile<4>(fT + (size_t)b * N_SP * MID_CH, MID_CH,
               gT + (size_t)b * N_SP * MID_CH, MID_CH, i0, j0, MID_CH, acc);

  int t = threadIdx.x, lane = t & 31, w = t >> 5;
  int wm = w >> 1, wn = w & 1, hh = lane >> 4, l16 = lane & 15;
#pragma unroll
  for (int fr = 0; fr < 2; ++fr) {
    int ib = i0 + wm * 32 + fr * 16 + hh * 8;
#pragma unroll
    for (int fc = 0; fc < 4; ++fc) {
      int j = j0 + wn * 64 + fc * 16 + l16;
#pragma unroll
      for (int r = 0; r < 8; ++r)
        z[(size_t)(ib + r) * N_SP + j] = acc[fr][fc][r];
    }
  }
}

// ---------------------------------------------------------------------------
// Row softmax with scale; converts f32 logits to bf16 probs IN PLACE
// (bf16 row packed at the head of each f32 row; pitch stays N_SP floats).
// ---------------------------------------------------------------------------
__global__ void softmax_rows(float* __restrict__ z)
{
  __shared__ float red[256];
  int i = blockIdx.x, t = threadIdx.x;
  float* row = z + (size_t)i * N_SP;
  float v[9];
  float m = -3.4e38f;
#pragma unroll
  for (int k = 0; k < 9; ++k) {
    v[k] = row[t + k * 256] * SM_SCALE;
    m = fmaxf(m, v[k]);
  }
  red[t] = m; __syncthreads();
  for (int s = 128; s > 0; s >>= 1) {
    if (t < s) red[t] = fmaxf(red[t], red[t + s]);
    __syncthreads();
  }
  m = red[0]; __syncthreads();
  float sum = 0.f;
#pragma unroll
  for (int k = 0; k < 9; ++k) { v[k] = __expf(v[k] - m); sum += v[k]; }
  red[t] = sum; __syncthreads();
  for (int s = 128; s > 0; s >>= 1) {
    if (t < s) red[t] += red[t + s];
    __syncthreads();
  }
  float inv = 1.0f / red[0];
  U16* rh = (U16*)row;   // all reads of this row completed before the barriers
#pragma unroll
  for (int k = 0; k < 9; ++k) rh[t + k * 256] = f2bf(v[k] * inv);
}

// ---------------------------------------------------------------------------
// out3[i][m] = sum_j attn[i][j] * h[m][j]  (one batch; out3 bf16 [b][n][512])
// attn rows are bf16 packed with pitch 2*N_SP halfs. Bt = h ([m][j], contig j).
// NF=2 (64-wide N tile) to keep 144 workgroups of parallelism.
// ---------------------------------------------------------------------------
__global__ void gemm_av(const float* __restrict__ zbuf, const U16* __restrict__ hb,
                        U16* __restrict__ out3, int b)
{
  int i0 = blockIdx.y * 128, m0 = blockIdx.x * 64;
  v8f acc[2][2];
  gemm_tile<2>((const U16*)zbuf, 2 * N_SP,
               hb + (size_t)b * MID_CH * N_SP, N_SP, i0, m0, N_SP, acc);

  int t = threadIdx.x, lane = t & 31, w = t >> 5;
  int wm = w >> 1, wn = w & 1, hh = lane >> 4, l16 = lane & 15;
#pragma unroll
  for (int fr = 0; fr < 2; ++fr) {
    int ib = i0 + wm * 32 + fr * 16 + hh * 8;
#pragma unroll
    for (int fc = 0; fc < 2; ++fc) {
      int m = m0 + wn * 32 + fc * 16 + l16;
#pragma unroll
      for (int r = 0; r < 8; ++r)
        out3[((size_t)b * N_SP + ib + r) * MID_CH + m] = f2bf(acc[fr][fc][r]);
    }
  }
}

// ---------------------------------------------------------------------------
// Final conv + bias + residual:
// out[b][c][n] = sum_m Wv[c][m] * out3[b][n][m] + bv[c] + x[b][c][n]
// ---------------------------------------------------------------------------
__global__ void gemm_out(const U16* __restrict__ Wvh, const U16* __restrict__ out3,
                         const float* __restrict__ bv, const float* __restrict__ xin,
                         float* __restrict__ out)
{
  int b = blockIdx.z;
  int c0 = blockIdx.y * 128, n0 = blockIdx.x * 128;
  v8f acc[2][4];
  gemm_tile<4>(Wvh, MID_CH, out3 + (size_t)b * N_SP * MID_CH, MID_CH, c0, n0, MID_CH, acc);

  int t = threadIdx.x, lane = t & 31, w = t >> 5;
  int wm = w >> 1, wn = w & 1, hh = lane >> 4, l16 = lane & 15;
#pragma unroll
  for (int fr = 0; fr < 2; ++fr) {
    int cb = c0 + wm * 32 + fr * 16 + hh * 8;
#pragma unroll
    for (int fc = 0; fc < 4; ++fc) {
      int n = n0 + wn * 64 + fc * 16 + l16;
#pragma unroll
      for (int r = 0; r < 8; ++r) {
        int c = cb + r;
        size_t idx = ((size_t)b * C_CH + c) * N_SP + n;
        out[idx] = acc[fr][fc][r] + bv[c] + xin[idx];
      }
    }
  }
}

// ---------------------------------------------------------------------------
extern "C" void kernel_launch(void* const* d_in, const int* in_sizes, int n_in,
                              void* d_out, int out_size, void* d_ws, size_t ws_size,
                              hipStream_t stream)
{
  const float* x     = (const float*)d_in[0];
  const float* Wf    = (const float*)d_in[1];
  const float* bf_   = (const float*)d_in[2];
  const float* gf    = (const float*)d_in[3];
  const float* betaf = (const float*)d_in[4];
  const float* mf    = (const float*)d_in[5];
  const float* vf    = (const float*)d_in[6];
  const float* Wg    = (const float*)d_in[7];
  const float* bg_   = (const float*)d_in[8];
  const float* gg    = (const float*)d_in[9];
  const float* betag = (const float*)d_in[10];
  const float* mg    = (const float*)d_in[11];
  const float* vg    = (const float*)d_in[12];
  const float* Wh    = (const float*)d_in[13];
  const float* bh_   = (const float*)d_in[14];
  const float* Wv    = (const float*)d_in[15];
  const float* bv    = (const float*)d_in[16];
  float* out = (float*)d_out;

  char* ws = (char*)d_ws;
  size_t off = 0;
  U16* Wq  = (U16*)(ws + off); off += (size_t)1536 * 1024 * 2;            // 3.0 MB
  U16* Wvh = (U16*)(ws + off); off += (size_t)1024 * 512 * 2;             // 1.0 MB
  U16* xT  = (U16*)(ws + off); off += (size_t)BATCH * N_SP * C_CH * 2;    // 75.5 MB
  U16* fT  = (U16*)(ws + off); off += (size_t)BATCH * N_SP * MID_CH * 2;  // 37.7 MB
  U16* gT  = (U16*)(ws + off); off += (size_t)BATCH * N_SP * MID_CH * 2;  // 37.7 MB
  U16* hb  = (U16*)(ws + off); off += (size_t)BATCH * MID_CH * N_SP * 2;  // 37.7 MB
  U16* o3  = (U16*)(ws + off); off += (size_t)BATCH * N_SP * MID_CH * 2;  // 37.7 MB
  float* z = (float*)(ws + off);                                          // 21.2 MB (1 batch)

  pack_w<<<8192, 256, 0, stream>>>(Wf, Wg, Wh, Wv, Wq, Wvh);
  pack_x<<<dim3(N_SP / 32, C_CH / 32, BATCH), 256, 0, stream>>>(x, xT);

  gemm_qkv<<<dim3(N_SP / 128, 1536 / 128, BATCH), 256, 0, stream>>>(
      Wq, xT, bf_, gf, betaf, mf, vf, bg_, gg, betag, mg, vg, bh_, fT, gT, hb);

  for (int b = 0; b < BATCH; ++b) {
    gemm_z<<<dim3(N_SP / 128, N_SP / 128), 256, 0, stream>>>(fT, gT, z, b);
    softmax_rows<<<N_SP, 256, 0, stream>>>(z);
    gemm_av<<<dim3(MID_CH / 64, N_SP / 128), 256, 0, stream>>>(z, hb, o3, b);
  }

  gemm_out<<<dim3(N_SP / 128, C_CH / 128, BATCH), 256, 0, stream>>>(Wvh, o3, bv, x, out);
}
